// BoundaryOffsetModule_90915867722253
// MI455X (gfx1250) — matrix operand order repaired and checked by reading, not compiled, not run on hardware
//
#include <hip/hip_runtime.h>
#include <hip/hip_bf16.h>

// ---------------------------------------------------------------------------
// EdgeConv-style boundary-offset module for MI455X (gfx1250, wave32).
// prep (weights->f16 n-major, softmax feat, packed (x,y,z,|p|^2))
//   -> KNN via v_wmma_f32_16x16x4_f32 score tiles + per-lane sorted top-16
//   -> per-point edge MLP via v_wmma_f32_16x16x32_f16 -> heads (WMMA).
// ---------------------------------------------------------------------------

typedef __attribute__((ext_vector_type(16))) _Float16 v16h;
typedef __attribute__((ext_vector_type(8)))  _Float16 v8h;
typedef __attribute__((ext_vector_type(8)))  float    v8f;
typedef __attribute__((ext_vector_type(2)))  float    v2f;

__device__ __forceinline__ v8f wmma16(v16h a, v16h b, v8f c) {
    // D = A(16x32 f16) x B(32x16 f16) + C(16x16 f32)
    return __builtin_amdgcn_wmma_f32_16x16x32_f16(
        false, a, false, b, (short)0, c, false, false);
}

__device__ __forceinline__ v8f wmma4(v2f a, v2f b, v8f c) {
    // D = A(16x4 f32) x B(4x16 f32) + C(16x16 f32)
    return __builtin_amdgcn_wmma_f32_16x16x4_f32(
        false, a, false, b, (short)0, c, false, false);
}

// A-fragment: 16x32 f16, row r = lane&15; lanes 0-15 hold K {0..7,16..23},
// lanes 16-31 hold K {8..15,24..31}  (ISA 7.12.2, 16-bit A layout).
__device__ __forceinline__ v16h load_frag_a(const _Float16* base, int ld, int k0) {
    int lane = threadIdx.x & 31;
    const _Float16* s = base + (size_t)(lane & 15) * ld + k0 + ((lane & 16) ? 8 : 0);
    v8h lo = *(const v8h*)(s);
    v8h hi = *(const v8h*)(s + 16);
    return __builtin_shufflevector(lo, hi, 0,1,2,3,4,5,6,7,8,9,10,11,12,13,14,15);
}

// B-fragment from an n-major (transposed) weight array Wt[n][k], ld = K:
// lane holds column n = lane&15; lanes 0-15: K {0..7,16..23}, lanes 16-31: K {8..15,24..31}.
__device__ __forceinline__ v16h load_frag_b(const _Float16* base, int ld, int k0) {
    int lane = threadIdx.x & 31;
    const _Float16* s = base + (size_t)(lane & 15) * ld + k0 + ((lane & 16) ? 8 : 0);
    v8h lo = *(const v8h*)(s);
    v8h hi = *(const v8h*)(s + 16);
    return __builtin_shufflevector(lo, hi, 0,1,2,3,4,5,6,7,8,9,10,11,12,13,14,15);
}

// ---------------------------------------------------------------------------
// Prep: transpose + pad weights to f16 n-major so B-fragments are contiguous.
// ---------------------------------------------------------------------------
__global__ void prep_weights_kernel(const float* __restrict__ W1,
                                    const float* __restrict__ W2,
                                    const float* __restrict__ W3,
                                    const float* __restrict__ W4,
                                    _Float16* __restrict__ wt1,
                                    _Float16* __restrict__ wt2,
                                    _Float16* __restrict__ wt3,
                                    _Float16* __restrict__ wt4) {
    int tid = threadIdx.x;
    for (int i = tid; i < 64 * 64; i += 256) {
        int n = i >> 6, k = i & 63;
        wt1[i] = (k < 35) ? (_Float16)W1[k * 64 + n] : (_Float16)0.f;  // W1: 35x64
        wt2[i] = (_Float16)W2[k * 64 + n];                             // W2: 64x64
    }
    for (int i = tid; i < 32 * 64; i += 256) {
        int n = i >> 6, k = i & 63;
        wt3[i] = (_Float16)W3[k * 32 + n];                             // W3: 64x32
    }
    for (int i = tid; i < 16 * 32; i += 256) {
        int n = i >> 5, k = i & 31;
        wt4[i] = (n < 3) ? (_Float16)W4[k * 3 + n] : (_Float16)0.f;    // W4: 32x3
    }
}

// feat[i] = softmax(seg_logits[i]) (13) ++ coord[i] (3), f32
// pj4[i]  = (x, y, z, |p|^2) packed for the WMMA KNN scorer.
__global__ void prep_feat_kernel(const float* __restrict__ seg,
                                 const float* __restrict__ coord,
                                 float* __restrict__ feat,
                                 float4* __restrict__ pj4, int N) {
    int i = blockIdx.x * blockDim.x + threadIdx.x;
    if (i >= N) return;
    float l[13];
    float m = -3.4e38f;
#pragma unroll
    for (int c = 0; c < 13; ++c) { l[c] = seg[i * 13 + c]; m = fmaxf(m, l[c]); }
    float s = 0.f;
#pragma unroll
    for (int c = 0; c < 13; ++c) { l[c] = __expf(l[c] - m); s += l[c]; }
    float inv = 1.f / s;
#pragma unroll
    for (int c = 0; c < 13; ++c) feat[(size_t)i * 16 + c] = l[c] * inv;
    float x = coord[i * 3 + 0], y = coord[i * 3 + 1], z = coord[i * 3 + 2];
    feat[(size_t)i * 16 + 13] = x;
    feat[(size_t)i * 16 + 14] = y;
    feat[(size_t)i * 16 + 15] = z;
    float4 p; p.x = x; p.y = y; p.z = z;
    p.w = fmaf(x, x, fmaf(y, y, z * z));
    pj4[i] = p;
}

// ---------------------------------------------------------------------------
// KNN via WMMA f32 16x16x4.
// Wave handles 16 query points i0..i0+15. Score tile:
//   D[j][i] = |p_j|^2 - 2 p_j . p_i   (missing |p_i|^2 is per-column constant
//                                      -> does not change per-query top-k)
// A row j = (x_j, y_j, z_j, |p_j|^2), B col i = (-2x_i, -2y_i, -2z_i, 1).
// D layout: lane holds query column i = lane&15; v8f elems = 8 candidate rows
// (lanes 0-15: rows 0-7, lanes 16-31: rows 8-15) -> each lane filters its own
// candidate stream into a sorted top-16 register list; lane pairs merge at end.
// ---------------------------------------------------------------------------
#define KCHUNK 256
__global__ __launch_bounds__(256) void knn_wmma_kernel(
    const float4* __restrict__ pj4, int* __restrict__ nbr, int N) {
    __shared__ __align__(16) float4 sc4[KCHUNK];
    const int tid = threadIdx.x, lane = tid & 31, w = tid >> 5;
    const int i0 = (blockIdx.x * 8 + w) * 16;
    const int i = i0 + (lane & 15);
    const bool act = (i0 < N);                 // wave-uniform

    // B fragment (constant): lanes 0-15 hold K{0,1}=(-2x,-2y), 16-31 K{2,3}=(-2z,1)
    v2f bfrag; bfrag[0] = 0.f; bfrag[1] = 0.f;
    if (i < N) {
        float4 p = pj4[i];
        bfrag[0] = (lane & 16) ? (-2.f * p.z) : (-2.f * p.x);
        bfrag[1] = (lane & 16) ? 1.f          : (-2.f * p.y);
    }

    float bd[16]; int bi_[16];
#pragma unroll
    for (int t = 0; t < 16; ++t) { bd[t] = 3.0e38f; bi_[t] = (i < N) ? i : 0; }

    const v8f z8 = {0.f, 0.f, 0.f, 0.f, 0.f, 0.f, 0.f, 0.f};
    const int rbase = (lane & 16) ? 8 : 0;

    for (int c0 = 0; c0 < N; c0 += KCHUNK) {
        __syncthreads();
        {   // cooperative stage of KCHUNK candidates (+ prefetch of next chunk)
            int j = c0 + tid;
            float4 v;
            if (j < N) v = pj4[j];
            else { v.x = 0.f; v.y = 0.f; v.z = 0.f; v.w = 3.0e38f; }
            sc4[tid] = v;
            int jn = j + KCHUNK;
            if (jn < N) __builtin_prefetch(&pj4[jn], 0, 0);
        }
        __syncthreads();

        if (act) {
#pragma unroll 2
            for (int t = 0; t < KCHUNK / 16; ++t) {
                // A fragment: row m = lane&15 of this 16-candidate tile
                const float* s = (const float*)&sc4[t * 16 + (lane & 15)] +
                                 ((lane & 16) ? 2 : 0);
                v2f afrag = *(const v2f*)s;
                v8f d = wmma4(afrag, bfrag, z8);
                int jrow = c0 + t * 16 + rbase;
#pragma unroll
                for (int r = 0; r < 8; ++r) {
                    float dv = d[r];
                    if (dv < bd[15]) {          // rare after warm-up
                        float dk = dv; int ik = jrow + r;
#pragma unroll
                        for (int u = 0; u < 16; ++u) {
                            bool sw = dk < bd[u];
                            float td = sw ? bd[u] : dk;
                            int   ti = sw ? bi_[u] : ik;
                            bd[u]  = sw ? dk : bd[u];
                            bi_[u] = sw ? ik : bi_[u];
                            dk = td; ik = ti;
                        }
                    }
                }
            }
        }
    }

    if (act) {
        // merge lane L (rows 0-7 stream) with lane L+16 (rows 8-15 stream):
        // snapshot partner's sorted list first, then gated-insert all 16.
        float pd[16]; int pi_[16];
#pragma unroll
        for (int t = 0; t < 16; ++t) {
            pd[t]  = __shfl_xor(bd[t], 16);
            pi_[t] = __shfl_xor(bi_[t], 16);
        }
#pragma unroll
        for (int t = 0; t < 16; ++t) {
            float dk = pd[t]; int ik = pi_[t];
            if (dk < bd[15]) {
#pragma unroll
                for (int u = 0; u < 16; ++u) {
                    bool sw = dk < bd[u];
                    float td = sw ? bd[u] : dk;
                    int   ti = sw ? bi_[u] : ik;
                    bd[u]  = sw ? dk : bd[u];
                    bi_[u] = sw ? ik : bi_[u];
                    dk = td; ik = ti;
                }
            }
        }
        if (lane < 16 && i < N) {
#pragma unroll
            for (int t = 0; t < 16; ++t) nbr[(size_t)i * 16 + t] = bi_[t];
        }
    }
}

// ---------------------------------------------------------------------------
// Edge MLP: block = 8 waves, wave w handles point p. 16 edges == 16-row tile.
// ---------------------------------------------------------------------------
__global__ __launch_bounds__(256) void edge_mlp_kernel(
    const float* __restrict__ feat, const int* __restrict__ nbr,
    const _Float16* __restrict__ gwt1, const _Float16* __restrict__ gwt2,
    const float* __restrict__ b1, const float* __restrict__ b2,
    _Float16* __restrict__ pooled, int N) {
    __shared__ __align__(16) _Float16 sW1[64 * 64];
    __shared__ __align__(16) _Float16 sW2[64 * 64];
    __shared__ __align__(16) _Float16 sET[8][16 * 64];
    __shared__ __align__(16) _Float16 sH1[8][16 * 64];
    const int tid = threadIdx.x, lane = tid & 31, w = tid >> 5;

    {   // cooperative weight staging (16B chunks)
        const uint4* s1 = (const uint4*)gwt1; uint4* d1 = (uint4*)sW1;
        const uint4* s2 = (const uint4*)gwt2; uint4* d2 = (uint4*)sW2;
        for (int k = tid; k < 512; k += 256) { d1[k] = s1[k]; d2[k] = s2[k]; }
    }

    const int p = blockIdx.x * 8 + w;
    const bool act = (p < N);

    if (act && lane < 16) {  // edge tile: [feat_c | feat_n - feat_c | cdiff | 0pad]
        int e = lane;
        int j = nbr[(size_t)p * 16 + e];
        const float* fp = feat + (size_t)p * 16;
        const float* fj = feat + (size_t)j * 16;
        _Float16* row = &sET[w][e * 64];
#pragma unroll
        for (int c = 0; c < 16; ++c) {
            float a = fp[c], b = fj[c];
            row[c]      = (_Float16)a;
            row[16 + c] = (_Float16)(b - a);
        }
#pragma unroll
        for (int k = 0; k < 3; ++k) row[32 + k] = (_Float16)(fj[13 + k] - fp[13 + k]);
#pragma unroll
        for (int c = 35; c < 64; ++c) row[c] = (_Float16)0.f;
    }
    __syncthreads();

    const v8f z8 = {0.f, 0.f, 0.f, 0.f, 0.f, 0.f, 0.f, 0.f};
    const int n = lane & 15, mb = (lane & 16) ? 8 : 0;

    if (act) {  // layer 1
        v16h a0 = load_frag_a(&sET[w][0], 64, 0);
        v16h a1 = load_frag_a(&sET[w][0], 64, 32);
        v8f acc[4] = {z8, z8, z8, z8};
#pragma unroll
        for (int t = 0; t < 4; ++t) {
            acc[t] = wmma16(a0, load_frag_b(sW1 + t * 16 * 64, 64, 0),  acc[t]);
            acc[t] = wmma16(a1, load_frag_b(sW1 + t * 16 * 64, 64, 32), acc[t]);
        }
#pragma unroll
        for (int t = 0; t < 4; ++t) {
            float bv = b1[t * 16 + n];
#pragma unroll
            for (int r = 0; r < 8; ++r)
                sH1[w][(mb + r) * 64 + t * 16 + n] = (_Float16)fmaxf(acc[t][r] + bv, 0.f);
        }
    }
    __syncthreads();

    if (act) {  // layer 2 + edge max-pool + relu
        v16h a0 = load_frag_a(&sH1[w][0], 64, 0);
        v16h a1 = load_frag_a(&sH1[w][0], 64, 32);
        v8f acc[4] = {z8, z8, z8, z8};
#pragma unroll
        for (int t = 0; t < 4; ++t) {
            acc[t] = wmma16(a0, load_frag_b(sW2 + t * 16 * 64, 64, 0),  acc[t]);
            acc[t] = wmma16(a1, load_frag_b(sW2 + t * 16 * 64, 64, 32), acc[t]);
        }
#pragma unroll
        for (int t = 0; t < 4; ++t) {
            float bv = b2[t * 16 + n];
            float pm = fmaxf(acc[t][0] + bv, 0.f);   // relu; pool of relu'd >= 0
#pragma unroll
            for (int r = 1; r < 8; ++r) pm = fmaxf(pm, acc[t][r] + bv);
            pm = fmaxf(pm, __shfl_xor(pm, 16));      // rows 0-7 vs 8-15
            if (lane < 16)
                pooled[(size_t)p * 64 + t * 16 + lane] = (_Float16)pm;
        }
    }
}

// ---------------------------------------------------------------------------
// Heads: wave handles 16 points. pooled(16x64) x Wt3(64x32) -> relu ->
// (16x32) x Wt4(32x16 pad) -> cols 0..2 + b4 -> out.
// ---------------------------------------------------------------------------
__global__ __launch_bounds__(256) void heads_kernel(
    const _Float16* __restrict__ pooled, const _Float16* __restrict__ gwt3,
    const _Float16* __restrict__ gwt4, const float* __restrict__ b3,
    const float* __restrict__ b4, float* __restrict__ out, int N) {
    __shared__ __align__(16) _Float16 sW3[32 * 64];
    __shared__ __align__(16) _Float16 sW4[16 * 32];
    __shared__ __align__(16) _Float16 sH3[8][16 * 32];
    const int tid = threadIdx.x, lane = tid & 31, w = tid >> 5;
    {
        const uint4* s3 = (const uint4*)gwt3; uint4* d3 = (uint4*)sW3;
        for (int k = tid; k < 256; k += 256) d3[k] = s3[k];
        const uint4* s4 = (const uint4*)gwt4; uint4* d4 = (uint4*)sW4;
        for (int k = tid; k < 64; k += 256) d4[k] = s4[k];
    }
    __syncthreads();

    const int pt = blockIdx.x * 8 + w;
    const int p0 = pt * 16;
    const bool act = (p0 < N);
    const v8f z8 = {0.f, 0.f, 0.f, 0.f, 0.f, 0.f, 0.f, 0.f};
    const int n = lane & 15, mb = (lane & 16) ? 8 : 0;

    if (act) {  // layer 3: 64 -> 32, relu
        const _Float16* ap = pooled + (size_t)p0 * 64;
        v16h a0 = load_frag_a(ap, 64, 0);
        v16h a1 = load_frag_a(ap, 64, 32);
        v8f acc[2] = {z8, z8};
#pragma unroll
        for (int t = 0; t < 2; ++t) {
            acc[t] = wmma16(a0, load_frag_b(sW3 + t * 16 * 64, 64, 0),  acc[t]);
            acc[t] = wmma16(a1, load_frag_b(sW3 + t * 16 * 64, 64, 32), acc[t]);
        }
#pragma unroll
        for (int t = 0; t < 2; ++t) {
            float bv = b3[t * 16 + n];
#pragma unroll
            for (int r = 0; r < 8; ++r)
                sH3[w][(mb + r) * 32 + t * 16 + n] = (_Float16)fmaxf(acc[t][r] + bv, 0.f);
        }
    }
    __syncthreads();

    if (act) {  // layer 4: 32 -> 3 (padded to 16 cols)
        v16h a = load_frag_a(&sH3[w][0], 32, 0);
        v16h b = load_frag_b(sW4, 32, 0);
        v8f c4 = z8;
        c4 = wmma16(a, b, c4);
        if (n < 3) {
            float bv = b4[n];
#pragma unroll
            for (int r = 0; r < 8; ++r) {
                int row = p0 + mb + r;
                if (row < N) out[(size_t)row * 3 + n] = c4[r] + bv;
            }
        }
    }
}

// ---------------------------------------------------------------------------
extern "C" void kernel_launch(void* const* d_in, const int* in_sizes, int n_in,
                              void* d_out, int out_size, void* d_ws, size_t ws_size,
                              hipStream_t stream) {
    const float* seg   = (const float*)d_in[0];
    const float* coord = (const float*)d_in[1];
    // d_in[2] = offset (single batch [N]) -> all points share a batch; unused.
    const float* W1 = (const float*)d_in[3];
    const float* b1 = (const float*)d_in[4];
    const float* W2 = (const float*)d_in[5];
    const float* b2 = (const float*)d_in[6];
    const float* W3 = (const float*)d_in[7];
    const float* b3 = (const float*)d_in[8];
    const float* W4 = (const float*)d_in[9];
    const float* b4 = (const float*)d_in[10];
    float* out = (float*)d_out;

    const int N = in_sizes[1] / 3;

    // workspace layout (all 16B-aligned); ~4.6 MB total for N=16384
    char* ws = (char*)d_ws;
    _Float16* wt1 = (_Float16*)(ws);             // 64*64*2 = 8192 B
    _Float16* wt2 = (_Float16*)(ws + 8192);      // 8192 B
    _Float16* wt3 = (_Float16*)(ws + 16384);     // 4096 B
    _Float16* wt4 = (_Float16*)(ws + 20480);     // 1024 B
    size_t off = 32768;
    float* feat = (float*)(ws + off);            // N*16*4
    off += (size_t)N * 16 * 4;
    float4* pj4 = (float4*)(ws + off);           // N*16 B  (x,y,z,|p|^2)
    off += (size_t)N * 16;
    int* nbr = (int*)(ws + off);                 // N*16*4
    off += (size_t)N * 16 * 4;
    _Float16* pooled = (_Float16*)(ws + off);    // N*64*2

    prep_weights_kernel<<<1, 256, 0, stream>>>(W1, W2, W3, W4, wt1, wt2, wt3, wt4);
    prep_feat_kernel<<<(N + 255) / 256, 256, 0, stream>>>(seg, coord, feat, pj4, N);

    const int itiles = (N + 15) / 16;
    knn_wmma_kernel<<<(itiles + 7) / 8, 256, 0, stream>>>(pj4, nbr, N);

    edge_mlp_kernel<<<(N + 7) / 8, 256, 0, stream>>>(feat, nbr, wt1, wt2, b1, b2, pooled, N);
    heads_kernel<<<(itiles + 7) / 8, 256, 0, stream>>>(pooled, wt3, wt4, b3, b4, out, N);
}
